// Attention_st_2010044694918
// MI455X (gfx1250) — matrix operand order repaired and checked by reading, not compiled, not run on hardware
//
#include <hip/hip_runtime.h>
#include <hip/hip_bf16.h>

// ---------------------------------------------------------------------------
// MI455X (gfx1250) fused attention + projection.
//   out = concat( proj( permute( softmax(Q Kt * s) @ K ) ), K )
// bf16 WMMA (v_wmma_f32_16x16x32_bf16) with f32 accumulation everywhere.
// All accumulators are single live v8f registers (no arrays -> no scratch).
// Reduction loops unrolled x4 for load/WMMA overlap.
// Workspace: B*64*768*4 = 48MB for the permuted O matrix.
// ---------------------------------------------------------------------------

typedef __bf16 bf16_t;
typedef bf16_t v16bf __attribute__((ext_vector_type(16)));
typedef float  v8f   __attribute__((ext_vector_type(8)));

constexpr int BATCH = 256;
constexpr int SEQ   = 320;
constexpr int CH    = 768;
constexpr int LQ    = 64;     // query rows
constexpr int LK    = 256;    // key rows
constexpr float SCALE = 0.125f;   // (768/12)^-0.5

// LDS: S (64x256 f32) + P (64x256 bf16) + reduction pad (4x64 f32)
constexpr int SMEM_S   = LQ * LK * 4;         // 65536
constexpr int SMEM_P   = LQ * LK * 2;         // 32768
constexpr int SMEM_RED = 4 * LQ * 4;          // 1024
constexpr int SMEM_BYTES = SMEM_S + SMEM_P + SMEM_RED;

// ---- WMMA wrapper ---------------------------------------------------------
__device__ __forceinline__ v8f wmma_bf16(v16bf a, v16bf b, v8f c) {
  // (neg_a, A, neg_b, B, c_mod, C, reuse_a, reuse_b)
  return __builtin_amdgcn_wmma_f32_16x16x32_bf16(false, a, false, b,
                                                 (short)0, c, false, false);
}

// ---- Fragment builders: contiguous k = half*16 + s for both A and B (not ISA table order) ----
// A 16x32 bf16: lane = row m (l&15); VGPR r -> K = half*16+2r (matches B)
__device__ __forceinline__ v16bf load_A_rowmajor_f32(const float* __restrict__ base,
                                                     int ld, int lane) {
  const int m = lane & 15, half = lane >> 4;
  v16bf a;
#pragma unroll
  for (int r = 0; r < 8; ++r) {
    const int k = half * 16 + 2 * r;
    const float2 f = *(const float2*)(base + (size_t)m * ld + k); // contiguous pair
    a[2 * r]     = (bf16_t)f.x;
    a[2 * r + 1] = (bf16_t)f.y;
  }
  return a;
}

// A fragment from LDS bf16 row-major (probabilities P)
__device__ __forceinline__ v16bf load_A_lds_bf16(const bf16_t* base, int ld, int lane) {
  const int m = lane & 15, half = lane >> 4;
  v16bf a;
#pragma unroll
  for (int r = 0; r < 8; ++r) {
    const int k = half * 16 + 2 * r;
    a[2 * r]     = base[m * ld + k];
    a[2 * r + 1] = base[m * ld + k + 1];
  }
  return a;
}

// B 32x16 bf16: lane = col n (l&15); lanes0-15 rows 0..15 (2/VGPR), lanes16-31 rows 16..31.
// Transposed source: B[k][n] = mem[n*ld + k]  (reduction dim contiguous -> float2)
__device__ __forceinline__ v16bf load_B_trans_f32(const float* __restrict__ base,
                                                  int ld, int lane) {
  const int n = lane & 15, khalf = lane >> 4;
  v16bf b;
#pragma unroll
  for (int r = 0; r < 8; ++r) {
    const int k = khalf * 16 + 2 * r;
    const float2 f = *(const float2*)(base + (size_t)n * ld + k);
    b[2 * r]     = (bf16_t)f.x;
    b[2 * r + 1] = (bf16_t)f.y;
  }
  return b;
}

// Row-major source: B[k][n] = mem[k*ld + n] (used for P@K).
__device__ __forceinline__ v16bf load_B_rowmajor_f32(const float* __restrict__ base,
                                                     int ld, int lane) {
  const int n = lane & 15, khalf = lane >> 4;
  v16bf b;
#pragma unroll
  for (int r = 0; r < 8; ++r) {
    const int k0 = khalf * 16 + 2 * r;
    b[2 * r]     = (bf16_t)base[(size_t)(k0)     * ld + n];
    b[2 * r + 1] = (bf16_t)base[(size_t)(k0 + 1) * ld + n];
  }
  return b;
}

// ---------------------------------------------------------------------------
// Kernel 1: per-batch attention.  1 block = 1 batch, 256 threads = 8 waves.
// ---------------------------------------------------------------------------
__global__ void attn_scores_softmax_pv(const float* __restrict__ x,
                                       float* __restrict__ o_ws) {
  extern __shared__ char lds[];
  float*  Sf  = (float*)lds;                       // 64 x 256 scores (f32)
  bf16_t* P   = (bf16_t*)(lds + SMEM_S);           // 64 x 256 probs  (bf16)
  float*  red = (float*)(lds + SMEM_S + SMEM_P);   // 4 x 64 reduce pad

  const int b     = blockIdx.x;
  const float* xb = x + (size_t)b * SEQ * CH;
  const float* Q  = xb;             // rows 0..63
  const float* K  = xb + LQ * CH;   // rows 64..319

  const int tid  = threadIdx.x;
  const int wave = tid >> 5;        // wave32 (gfx1250)
  const int lane = tid & 31;
  const int mt   = wave & 3;        // M tile: rows mt*16..
  const int nh   = wave >> 2;       // column half: cols nh*128..

  // ---- Pass 1: S = scale * Q @ K^T.  One S tile at a time: single live acc.
  const float* Abase = Q + (size_t)mt * 16 * CH;
#pragma unroll 1
  for (int t = 0; t < 8; ++t) {
    const int n0 = nh * 128 + t * 16;
    const float* Bbase = K + (size_t)n0 * CH;
    v8f acc = (v8f){};
#pragma unroll 4
    for (int c0 = 0; c0 < CH; c0 += 32) {
      const v16bf a  = load_A_rowmajor_f32(Abase + c0, CH, lane);  // Q tile (L2-hot)
      const v16bf bb = load_B_trans_f32(Bbase + c0, CH, lane);     // K rows
      acc = wmma_bf16(a, bb, acc);
    }
    // C/D layout: VGPR v, lane l -> row (l>>4)*8+v, col l&15
    const int mrow = mt * 16 + (lane >> 4) * 8;
    const int n    = n0 + (lane & 15);
#pragma unroll
    for (int v = 0; v < 8; ++v) Sf[(mrow + v) * LK + n] = acc[v] * SCALE;
  }
  __syncthreads();

  // ---- Softmax: 4 threads per row, LDS tree reduction --------------------
  {
    const int row = tid & 63;        // 0..63
    const int q   = tid >> 6;        // 0..3 quarter of the 256 cols
    float* srow = Sf + row * LK + q * 64;

    float mx = -3.0e38f;
#pragma unroll 4
    for (int j = 0; j < 64; ++j) mx = fmaxf(mx, srow[j]);
    red[q * LQ + row] = mx;
    __syncthreads();
    mx = fmaxf(fmaxf(red[row], red[LQ + row]),
               fmaxf(red[2 * LQ + row], red[3 * LQ + row]));

    float sum = 0.f;
#pragma unroll 4
    for (int j = 0; j < 64; ++j) {
      const float e = __expf(srow[j] - mx);
      srow[j] = e;
      sum += e;
    }
    __syncthreads();                 // everyone done reading maxes
    red[q * LQ + row] = sum;
    __syncthreads();
    const float inv = 1.f / (red[row] + red[LQ + row] +
                             red[2 * LQ + row] + red[3 * LQ + row]);
    bf16_t* prow = P + row * LK + q * 64;
#pragma unroll 4
    for (int j = 0; j < 64; ++j) prow[j] = (bf16_t)(srow[j] * inv);
  }
  __syncthreads();

  // ---- Pass 2: O = P @ K, stored with the transpose-reshape permutation:
  // flat(O2) = n*64 + m  (O2 = reshape(O^T)).
  float* ob = o_ws + (size_t)b * LQ * CH;
  const bf16_t* Pbase = P + mt * 16 * LK;
#pragma unroll 1
  for (int t = 0; t < 24; ++t) {
    const int ct = (t * 2 + nh) * 16;            // channel tile base
    v8f oacc = (v8f){};
#pragma unroll 4
    for (int k0 = 0; k0 < LK; k0 += 32) {
      const v16bf a  = load_A_lds_bf16(Pbase + k0, LK, lane);
      const v16bf bb = load_B_rowmajor_f32(K + (size_t)k0 * CH + ct, CH, lane);
      oacc = wmma_bf16(a, bb, oacc);
    }
    const int n = ct + (lane & 15);
#pragma unroll
    for (int v = 0; v < 8; ++v) {
      const int m = mt * 16 + (lane >> 4) * 8 + v;
      ob[n * LQ + m] = oacc[v];                  // permuted store (flat = n*64+m)
    }
  }
}

// ---------------------------------------------------------------------------
// Kernel 2: Y = O2 @ W^T + bias -> output rows 0..63 of each batch.
// ---------------------------------------------------------------------------
__global__ void proj_kernel(const float* __restrict__ o_ws,
                            const float* __restrict__ w,
                            const float* __restrict__ bias,
                            float* __restrict__ out) {
  const int b = blockIdx.x;
  const float* O2 = o_ws + (size_t)b * LQ * CH;  // 64 x 768 row-major
  float* yb = out + (size_t)b * SEQ * CH;        // rows 0..63

  const int tid = threadIdx.x, wave = tid >> 5, lane = tid & 31;
  const int mt = wave & 3, nh = wave >> 2;
  const float* Abase = O2 + (size_t)mt * 16 * CH;

#pragma unroll 1
  for (int t = 0; t < 24; ++t) {
    const int o0 = (t * 2 + nh) * 16;
    const float* Bbase = w + (size_t)o0 * CH;    // B[j][o] = w[(o0+o)*CH + j]
    v8f acc = (v8f){};
#pragma unroll 4
    for (int j0 = 0; j0 < CH; j0 += 32) {
      const v16bf a  = load_A_rowmajor_f32(Abase + j0, CH, lane);
      const v16bf bb = load_B_trans_f32(Bbase + j0, CH, lane);
      acc = wmma_bf16(a, bb, acc);
    }
    const int o    = o0 + (lane & 15);
    const float bv = bias[o];
#pragma unroll
    for (int v = 0; v < 8; ++v) {
      const int m = mt * 16 + (lane >> 4) * 8 + v;
      yb[(size_t)m * CH + o] = acc[v] + bv;
    }
  }
}

// ---------------------------------------------------------------------------
// Kernel 3: concat K -> output rows 64..319 == identity copy of those rows.
// 2D grid: blockIdx.y = batch, no div/mod.  float4 streaming.
// ---------------------------------------------------------------------------
__global__ void copy_k_kernel(const float* __restrict__ x, float* __restrict__ out) {
  const size_t b   = blockIdx.y;
  const size_t off = b * ((size_t)SEQ * CH / 4) + (size_t)LQ * CH / 4
                   + (size_t)blockIdx.x * blockDim.x + threadIdx.x;
  ((float4*)out)[off] = ((const float4*)x)[off];
}

// ---------------------------------------------------------------------------
extern "C" void kernel_launch(void* const* d_in, const int* in_sizes, int n_in,
                              void* d_out, int out_size, void* d_ws, size_t ws_size,
                              hipStream_t stream) {
  const float* x    = (const float*)d_in[0];   // (256, 320, 768) f32
  const float* w    = (const float*)d_in[1];   // (768, 768) f32
  const float* bias = (const float*)d_in[2];   // (768,) f32
  float* out  = (float*)d_out;                 // (256, 320, 768) f32
  float* o_ws = (float*)d_ws;                  // needs 48MB

  attn_scores_softmax_pv<<<BATCH, 256, SMEM_BYTES, stream>>>(x, o_ws);
  proj_kernel<<<BATCH, 256, 0, stream>>>(o_ws, w, bias, out);
  // (SEQ-LQ)*CH/4 = 49152 float4s per batch = 192 blocks of 256 threads
  copy_k_kernel<<<dim3(192, BATCH), 256, 0, stream>>>(x, out);
}
